// GATNet_45981919871651
// MI455X (gfx1250) — matrix operand order repaired and checked by reading, not compiled, not run on hardware
//
#include <hip/hip_runtime.h>
#include <math.h>

typedef _Float16 half_t;
typedef __attribute__((ext_vector_type(16))) _Float16 v16h;
typedef __attribute__((ext_vector_type(8)))  _Float16 v8h;
typedef __attribute__((ext_vector_type(4)))  _Float16 v4h;
typedef __attribute__((ext_vector_type(8)))  float    v8f;

static constexpr int N_NODES  = 10240;
static constexpr int N_EDGES  = 40960;
static constexpr int N_GRAPHS = 256;
static constexpr int NODES_PER_GRAPH = 40;   // 10240 / 256, batch = repeat(arange)

// ---------------------------------------------------------------------------
// Conversion / padding helpers (zero-padding K dims keeps GEMMs exact while
// letting every K tile hit the vectorized fast path)
// ---------------------------------------------------------------------------
__global__ void k_cvt_f16(const float* __restrict__ src, half_t* __restrict__ dst, long n) {
    for (long i = blockIdx.x * (long)blockDim.x + threadIdx.x; i < n;
         i += (long)gridDim.x * blockDim.x)
        dst[i] = (half_t)src[i];
}

// dst[r, 0:ldd] = { src[r, c] for c<C ; 0 for C<=c<ldd }
__global__ void k_cvt_pad_cols(const float* __restrict__ src, half_t* __restrict__ dst,
                               long rows, int C, int ldd) {
    long total = rows * ldd;
    for (long i = blockIdx.x * (long)blockDim.x + threadIdx.x; i < total;
         i += (long)gridDim.x * blockDim.x) {
        long r = i / ldd; int c = (int)(i % ldd);
        dst[i] = (c < C) ? (half_t)src[r * C + c] : (half_t)0.f;
    }
}

// dst[r, c] = { src[r, c] for r<R ; 0 for R<=r<Rpad }
__global__ void k_cvt_pad_rows(const float* __restrict__ src, half_t* __restrict__ dst,
                               int R, int C, int Rpad) {
    long total = (long)Rpad * C;
    for (long i = blockIdx.x * (long)blockDim.x + threadIdx.x; i < total;
         i += (long)gridDim.x * blockDim.x) {
        long r = i / C;
        dst[i] = (r < R) ? (half_t)src[i] : (half_t)0.f;
    }
}

// conv weight reorder: w[o, c, k] (OIH) -> wr[(k*C + c), o]
__global__ void k_conv_w_reorder(const float* __restrict__ w, half_t* __restrict__ wr,
                                 int O, int C, int Ks) {
    long total = (long)O * C * Ks;
    for (long i = blockIdx.x * (long)blockDim.x + threadIdx.x; i < total;
         i += (long)gridDim.x * blockDim.x) {
        int o = (int)(i / (C * Ks));
        int r = (int)(i % (C * Ks));
        int c = r / Ks, k = r % Ks;
        wr[(long)(k * C + c) * O + o] = (half_t)w[i];
    }
}

// embedding, channel-last for conv1: e16[b, m(0..127), s(0..999)] = emb[target[b,s], m]
__global__ void k_embed(const int* __restrict__ tgt, const float* __restrict__ emb,
                        half_t* __restrict__ e16) {
    const long total = (long)N_GRAPHS * 128 * 1000;
    for (long i = blockIdx.x * (long)blockDim.x + threadIdx.x; i < total;
         i += (long)gridDim.x * blockDim.x) {
        long b = i / 128000;
        int  r = (int)(i % 128000);
        int  m = r / 1000, s = r % 1000;
        int  t = tgt[b * 1000 + s];
        e16[i] = (half_t)emb[t * 128 + m];
    }
}

// ---------------------------------------------------------------------------
// WMMA GEMM: C[b] = act(A[b] @ B + bias).  A f16 with arbitrary row stride
// (so conv1d on channel-last data is this same GEMM with overlapping rows).
// Block: 256 threads = 8 wave32; block tile 64Mx64N; each wave: 2 x (16x16).
// Double-buffered LDS software pipeline: one barrier per K tile, next-tile
// global loads overlap the WMMAs.  B tile stored transposed so every fragment
// gather is two ds_load_b128.
// ---------------------------------------------------------------------------
#define LPITCH 40

__global__ __launch_bounds__(256) void k_wmma_gemm(
    const half_t* __restrict__ A, const half_t* __restrict__ B,
    const float* __restrict__ bias, float* __restrict__ C32, half_t* __restrict__ C16,
    int M, int N, int K, int lda, long sAb, int ldb, int ldc, long sCb, int act)
{
    __shared__ half_t lA [2][64 * LPITCH];   // [buf][row][k]
    __shared__ half_t lBt[2][64 * LPITCH];   // [buf][col][k]  (transposed)

    const int tid  = threadIdx.x;
    const int lane = tid & 31;
    const int wave = tid >> 5;
    const int mi   = wave & 3;            // M sub-tile 0..3
    const int nig  = wave >> 2;           // N group 0..1
    const int r16  = lane & 15;
    const int hi   = lane >> 4;
    const int tileM = blockIdx.y * 64;
    const int tileN = blockIdx.x * 64;
    const long aOff = (long)blockIdx.z * sAb;
    const long cOff = (long)blockIdx.z * sCb;

    // staging assignments (one b128 A load, two b64 B loads per thread)
    const int ar = tid >> 2;              // 0..63
    const int ac = (tid & 3) << 3;        // 0,8,16,24
    const int bk = tid >> 3;              // 0..31
    const int bn = (tid & 7) << 3;        // 0..56

    const bool aRowsIn = (tileM + 64 <= M);
    const bool bColsIn = (tileN + 64 <= N);

    // ---- tile loaders (fast vector path / branch-free clamped path) ----
    auto loadA = [&](int kk) -> v8h {
        if (aRowsIn && kk + 32 <= K)
            return *(const v8h*)(A + aOff + (long)(tileM + ar) * lda + kk + ac);
        v8h r = {};
        const int gr = tileM + ar;
        const int sr = gr < M ? gr : 0;
#pragma unroll
        for (int j = 0; j < 8; ++j) {
            int gc = kk + ac + j;
            int sc = gc < K ? gc : 0;
            half_t tv = A[aOff + (long)sr * lda + sc];          // clamped, branch-free
            r[j] = (gr < M && gc < K) ? tv : (half_t)0.f;
        }
        return r;
    };
    auto loadB = [&](int kk) -> v8h {
        if (bColsIn && kk + 32 <= K) {
            const half_t* bp = B + (long)(kk + bk) * ldb + tileN + bn;
            v4h b0 = *(const v4h*)bp;
            v4h b1 = *(const v4h*)(bp + 4);
            return __builtin_shufflevector(b0, b1, 0, 1, 2, 3, 4, 5, 6, 7);
        }
        v8h r = {};
        const int gk = kk + bk;
        const int sk = gk < K ? gk : 0;
#pragma unroll
        for (int j = 0; j < 8; ++j) {
            int gn = tileN + bn + j;
            int sn = gn < N ? gn : 0;
            half_t tv = B[(long)sk * ldb + sn];
            r[j] = (gk < K && gn < N) ? tv : (half_t)0.f;
        }
        return r;
    };
    auto storeA = [&](v8h v, int buf) {
        *(v8h*)(&lA[buf][ar * LPITCH + ac]) = v;
    };
    auto storeB = [&](v8h v, int buf) {           // transpose scatter into lBt[col][k]
#pragma unroll
        for (int j = 0; j < 8; ++j) lBt[buf][(bn + j) * LPITCH + bk] = v[j];
    };

    v8f acc0 = {0.f, 0.f, 0.f, 0.f, 0.f, 0.f, 0.f, 0.f};
    v8f acc1 = {0.f, 0.f, 0.f, 0.f, 0.f, 0.f, 0.f, 0.f};

    // ---- prologue: stage tile 0 ----
    {
        v8h pa = loadA(0);
        v8h pb = loadB(0);
        storeA(pa, 0);
        storeB(pb, 0);
    }
    __syncthreads();

    const int nTiles = (K + 31) >> 5;
    for (int t = 0; t < nTiles; ++t) {
        const int cur = t & 1;
        // ---- fragment gathers from current buffer: two ds_load_b128 each ----
        const v8h* lA8 = (const v8h*)(&lA[cur][0]);    // pitch = 5 v8h
        const v8h* lB8 = (const v8h*)(&lBt[cur][0]);
        const int arow = mi * 16 + r16;
        v8h alo = lA8[arow * 5 + 2 * hi];              // k = 16*hi + 0..7
        v8h ahi = lA8[arow * 5 + 2 * hi + 1];          // k = 16*hi + 8..15
        v16h af = __builtin_shufflevector(alo, ahi, 0,1,2,3,4,5,6,7,8,9,10,11,12,13,14,15);
        const int col0 = nig * 32 + r16;
        v8h b0lo = lB8[col0 * 5 + 2 * hi];             // k = 16*hi + 0..7
        v8h b0hi = lB8[col0 * 5 + 2 * hi + 1];         // k = 16*hi + 8..15
        v16h bf0 = __builtin_shufflevector(b0lo, b0hi, 0,1,2,3,4,5,6,7,8,9,10,11,12,13,14,15);
        const int col1 = col0 + 16;
        v8h b1lo = lB8[col1 * 5 + 2 * hi];
        v8h b1hi = lB8[col1 * 5 + 2 * hi + 1];
        v16h bf1 = __builtin_shufflevector(b1lo, b1hi, 0,1,2,3,4,5,6,7,8,9,10,11,12,13,14,15);

        // ---- issue next-tile global loads (overlap with WMMAs) ----
        const bool hasNext = (t + 1) < nTiles;
        v8h na = {}, nb = {};
        if (hasNext) {
            na = loadA((t + 1) << 5);
            nb = loadB((t + 1) << 5);
        }

        acc0 = __builtin_amdgcn_wmma_f32_16x16x32_f16(false, af, false, bf0,
                                                      (short)0, acc0, false, false);
        acc1 = __builtin_amdgcn_wmma_f32_16x16x32_f16(false, af, false, bf1,
                                                      (short)0, acc1, false, false);

        if (hasNext) {
            storeA(na, cur ^ 1);
            storeB(nb, cur ^ 1);
        }
        __syncthreads();
    }

    // ---- store (C/D layout: lane -> n = r16, vgpr v -> m = v + 8*hi) ----
    auto store_tile = [&](v8f a, int n) {
        if (n < N) {
            float bv = bias ? bias[n] : 0.f;
#pragma unroll
            for (int vv = 0; vv < 8; ++vv) {
                int m = tileM + mi * 16 + vv + (hi << 3);
                if (m < M) {
                    float val = a[vv] + bv;
                    if (act == 1) val = fmaxf(val, 0.f);
                    long idx = cOff + (long)m * ldc + n;
                    if (C32) C32[idx] = val;
                    if (C16) C16[idx] = (half_t)val;
                }
            }
        }
    };
    store_tile(acc0, tileN + nig * 32 + r16);
    store_tile(acc1, tileN + nig * 32 + 16 + r16);
}

// ---------------------------------------------------------------------------
// GAT attention machinery
// ---------------------------------------------------------------------------
__device__ inline unsigned fkey(float v) {
    unsigned u = __float_as_uint(v);
    return (u & 0x80000000u) ? ~u : (u | 0x80000000u);
}
__device__ inline float funkey(unsigned k) {
    return __uint_as_float((k & 0x80000000u) ? (k ^ 0x80000000u) : ~k);
}
__device__ inline float leaky(float v) { return v > 0.f ? v : 0.2f * v; }

__global__ void k_att_scores(const float* __restrict__ H, const float* __restrict__ aw_s,
                             const float* __restrict__ aw_d, float* __restrict__ oS,
                             float* __restrict__ oD, int heads, int F) {
    int id = blockIdx.x * blockDim.x + threadIdx.x;
    if (id >= N_NODES * heads) return;
    int n = id / heads, h = id % heads;
    const float* hp = H + (long)n * heads * F + (long)h * F;
    const float* sp = aw_s + h * F;
    const float* dp = aw_d + h * F;
    float ss = 0.f, dd = 0.f;
    for (int f = 0; f < F; ++f) { float x = hp[f]; ss += x * sp[f]; dd += x * dp[f]; }
    oS[id] = ss; oD[id] = dd;
}

__global__ void k_init_maxkey(unsigned* __restrict__ mk, int n) {
    int id = blockIdx.x * blockDim.x + threadIdx.x;
    if (id < n) mk[id] = 0x007FFFFFu;   // key(-inf)
}

__global__ void k_edge_max(const int* __restrict__ src, const int* __restrict__ dst,
                           const float* __restrict__ aS, const float* __restrict__ aD,
                           unsigned* __restrict__ mk, int heads) {
    const int EE = N_EDGES + N_NODES;
    long total = (long)EE * heads;
    for (long i = blockIdx.x * (long)blockDim.x + threadIdx.x; i < total;
         i += (long)gridDim.x * blockDim.x) {
        int e = (int)(i / heads), h = (int)(i % heads);
        int s = e < N_EDGES ? src[e] : e - N_EDGES;
        int d = e < N_EDGES ? dst[e] : e - N_EDGES;
        float v = leaky(aS[s * heads + h] + aD[d * heads + h]);
        atomicMax(&mk[d * heads + h], fkey(v));
    }
}

__global__ void k_edge_exp(const int* __restrict__ src, const int* __restrict__ dst,
                           const float* __restrict__ aS, const float* __restrict__ aD,
                           const unsigned* __restrict__ mk, float* __restrict__ ee,
                           float* __restrict__ denom, int heads) {
    const int EE = N_EDGES + N_NODES;
    long total = (long)EE * heads;
    for (long i = blockIdx.x * (long)blockDim.x + threadIdx.x; i < total;
         i += (long)gridDim.x * blockDim.x) {
        int e = (int)(i / heads), h = (int)(i % heads);
        int s = e < N_EDGES ? src[e] : e - N_EDGES;
        int d = e < N_EDGES ? dst[e] : e - N_EDGES;
        float v  = leaky(aS[s * heads + h] + aD[d * heads + h]);
        float ex = expf(v - funkey(mk[d * heads + h]));
        ee[i] = ex;
        atomicAdd(&denom[d * heads + h], ex);
    }
}

__global__ void k_edge_norm(float* __restrict__ ee, const float* __restrict__ denom,
                            const int* __restrict__ dst, int heads) {
    const int EE = N_EDGES + N_NODES;
    long total = (long)EE * heads;
    for (long i = blockIdx.x * (long)blockDim.x + threadIdx.x; i < total;
         i += (long)gridDim.x * blockDim.x) {
        int e = (int)(i / heads), h = (int)(i % heads);
        int d = e < N_EDGES ? dst[e] : e - N_EDGES;
        ee[i] = ee[i] / (denom[d * heads + h] + 1e-16f);
    }
}

__global__ void k_edge_aggr(const float* __restrict__ ee, const float* __restrict__ H,
                            float* __restrict__ OUT, const int* __restrict__ src,
                            const int* __restrict__ dst, int heads, int F) {
    const int EE = N_EDGES + N_NODES;
    const long HF = (long)heads * F;
    long total = (long)EE * HF;
    for (long i = blockIdx.x * (long)blockDim.x + threadIdx.x; i < total;
         i += (long)gridDim.x * blockDim.x) {
        int e = (int)(i / HF);
        int c = (int)(i % HF);
        int h = c / F;
        int s = e < N_EDGES ? src[e] : e - N_EDGES;
        int d = e < N_EDGES ? dst[e] : e - N_EDGES;
        atomicAdd(&OUT[(long)d * HF + c], ee[(long)e * heads + h] * H[(long)s * HF + c]);
    }
}

// bias + activation (1=relu, 2=elu); writes f16 with column padding to ldY
__global__ void k_bias_act16(const float* __restrict__ X, const float* __restrict__ bias,
                             half_t* __restrict__ Y, long rows, int C, int ldY, int act) {
    long total = rows * ldY;
    for (long i = blockIdx.x * (long)blockDim.x + threadIdx.x; i < total;
         i += (long)gridDim.x * blockDim.x) {
        long r = i / ldY; int c = (int)(i % ldY);
        float v = 0.f;
        if (c < C) {
            v = X[r * C + c] + bias[c];
            if (act == 1) v = v > 0.f ? v : 0.f;
            else if (act == 2) v = v > 0.f ? v : (expf(v) - 1.f);
        }
        Y[i] = (half_t)v;
    }
}

// graph global max pool (40 contiguous nodes); relu(max(x)+b) == max(relu(x+b))
__global__ void k_pool_graph(const float* __restrict__ OUT3, const float* __restrict__ b3,
                             half_t* __restrict__ g16, int C, int ldOut) {
    long total = (long)N_GRAPHS * ldOut;
    for (long i = blockIdx.x * (long)blockDim.x + threadIdx.x; i < total;
         i += (long)gridDim.x * blockDim.x) {
        int b = (int)(i / ldOut), c = (int)(i % ldOut);
        float m = 0.f;
        if (c < C) {
            m = -3.4e38f;
            for (int n = 0; n < NODES_PER_GRAPH; ++n)
                m = fmaxf(m, OUT3[((long)(b * NODES_PER_GRAPH + n)) * C + c]);
            m += b3[c];
            m = m > 0.f ? m : 0.f;
        }
        g16[i] = (half_t)m;
    }
}

// maxpool k=3 over spatial + flatten to reference order (ch*33 + pos)
__global__ void k_mpool_flatten(const half_t* __restrict__ c4, half_t* __restrict__ xt) {
    long total = (long)N_GRAPHS * 128 * 33;
    for (long i = blockIdx.x * (long)blockDim.x + threadIdx.x; i < total;
         i += (long)gridDim.x * blockDim.x) {
        int b = (int)(i / (128 * 33));
        int r = (int)(i % (128 * 33));
        int o = r / 33, q = r % 33;
        float m = -3.4e38f;
        for (int rr = 0; rr < 3; ++rr)
            m = fmaxf(m, (float)c4[(long)b * 12800 + (3 * q + rr) * 128 + o]);
        xt[i] = (half_t)m;   // i == b*4224 + o*33 + q
    }
}

__global__ void k_final(const half_t* __restrict__ a2, const float* __restrict__ w,
                        const float* __restrict__ bb, float* __restrict__ out) {
    int b = blockIdx.x * blockDim.x + threadIdx.x;
    if (b >= N_GRAPHS) return;
    float s = bb[0];
    for (int c = 0; c < 256; ++c) s += (float)a2[b * 256 + c] * w[c];
    out[b] = s;
}

// ---------------------------------------------------------------------------
// Host orchestration
// ---------------------------------------------------------------------------
static inline void gemm(hipStream_t stream, const half_t* A, const half_t* B,
                        const float* bias, float* C32, half_t* C16,
                        int M, int N, int K, int lda, long sAb, int ldb,
                        int ldc, long sCb, int act, int batch) {
    dim3 g((N + 63) / 64, (M + 63) / 64, batch);
    k_wmma_gemm<<<g, 256, 0, stream>>>(A, B, bias, C32, C16, M, N, K, lda, sAb, ldb, ldc, sCb, act);
}

static inline unsigned blk(long n) { return (unsigned)((n + 255) / 256); }

extern "C" void kernel_launch(void* const* d_in, const int* in_sizes, int n_in,
                              void* d_out, int out_size, void* d_ws, size_t ws_size,
                              hipStream_t stream) {
    (void)in_sizes; (void)n_in; (void)out_size; (void)ws_size;
    const float* x     = (const float*)d_in[0];
    const int*   eidx  = (const int*)d_in[1];
    const int*   tgt   = (const int*)d_in[3];
    const float* W1    = (const float*)d_in[4];
    const float* as1   = (const float*)d_in[5];
    const float* ad1   = (const float*)d_in[6];
    const float* b1    = (const float*)d_in[7];
    const float* W2    = (const float*)d_in[8];
    const float* as2   = (const float*)d_in[9];
    const float* ad2   = (const float*)d_in[10];
    const float* b2    = (const float*)d_in[11];
    const float* W3    = (const float*)d_in[12];
    const float* as3   = (const float*)d_in[13];
    const float* ad3   = (const float*)d_in[14];
    const float* b3    = (const float*)d_in[15];
    const float* fcg1w = (const float*)d_in[16];
    const float* fcg1b = (const float*)d_in[17];
    const float* fcg2w = (const float*)d_in[18];
    const float* fcg2b = (const float*)d_in[19];
    const float* emb   = (const float*)d_in[20];
    const float* cw1   = (const float*)d_in[21]; const float* cb1 = (const float*)d_in[22];
    const float* cw2   = (const float*)d_in[23]; const float* cb2 = (const float*)d_in[24];
    const float* cw3   = (const float*)d_in[25]; const float* cb3 = (const float*)d_in[26];
    const float* cw4   = (const float*)d_in[27]; const float* cb4 = (const float*)d_in[28];
    const float* fc1xtw = (const float*)d_in[29]; const float* fc1xtb = (const float*)d_in[30];
    const float* fc2xtw = (const float*)d_in[31]; const float* fc2xtb = (const float*)d_in[32];
    const float* fc1w  = (const float*)d_in[33]; const float* fc1b = (const float*)d_in[34];
    const float* fc2w  = (const float*)d_in[35]; const float* fc2b = (const float*)d_in[36];
    const float* outw  = (const float*)d_in[37]; const float* outb = (const float*)d_in[38];
    float* out = (float*)d_out;
    const int* srcIdx = eidx;
    const int* dstIdx = eidx + N_EDGES;

    // ---- workspace arena ----
    char* p = (char*)d_ws;
    auto alloc = [&](size_t n) { char* r = p; p += (n + 255) & ~(size_t)255; return r; };

    half_t* X0h   = (half_t*)alloc((size_t)N_NODES * 80 * 2);        // x padded 78->80
    half_t* W1h   = (half_t*)alloc((size_t)80 * 780 * 2);            // rows padded
    half_t* W2h   = (half_t*)alloc((size_t)800 * 1560 * 2);
    half_t* W3h   = (half_t*)alloc((size_t)1568 * 3120 * 2);
    half_t* g1wh  = (half_t*)alloc((size_t)3136 * 1024 * 2);
    half_t* g2wh  = (half_t*)alloc((size_t)1024 * 128 * 2);
    half_t* xt1wh = (half_t*)alloc((size_t)4224 * 1024 * 2);
    half_t* xt2wh = (half_t*)alloc((size_t)1024 * 128 * 2);
    half_t* f1wh  = (half_t*)alloc((size_t)256 * 1024 * 2);
    half_t* f2wh  = (half_t*)alloc((size_t)1024 * 256 * 2);
    half_t* cw1h  = (half_t*)alloc((size_t)8000 * 32 * 2);
    half_t* cw2h  = (half_t*)alloc((size_t)256 * 64 * 2);
    half_t* cw3h  = (half_t*)alloc((size_t)512 * 96 * 2);
    half_t* cw4h  = (half_t*)alloc((size_t)768 * 128 * 2);
    float*  aS    = (float*)alloc((size_t)N_NODES * 10 * 4);
    float*  aD    = (float*)alloc((size_t)N_NODES * 10 * 4);
    unsigned* mk  = (unsigned*)alloc((size_t)N_NODES * 10 * 4);
    float*  denom = (float*)alloc((size_t)N_NODES * 10 * 4);
    float*  ee    = (float*)alloc((size_t)(N_EDGES + N_NODES) * 10 * 4);
    half_t* gmax16 = (half_t*)alloc((size_t)N_GRAPHS * 3136 * 2);    // padded 3120->3136
    half_t* g1o   = (half_t*)alloc((size_t)N_GRAPHS * 1024 * 2);
    half_t* xc    = (half_t*)alloc((size_t)N_GRAPHS * 256 * 2);
    half_t* xt16  = (half_t*)alloc((size_t)N_GRAPHS * 4224 * 2);
    half_t* xto1  = (half_t*)alloc((size_t)N_GRAPHS * 1024 * 2);
    half_t* a1    = (half_t*)alloc((size_t)N_GRAPHS * 1024 * 2);
    half_t* a2    = (half_t*)alloc((size_t)N_GRAPHS * 256 * 2);
    char* regA = alloc((size_t)N_NODES * 3120 * 4);   // H1/H2/H3, later e16
    char* regB = alloc((size_t)N_NODES * 3120 * 4);   // OUT1/2/3, later conv outputs
    char* regC = alloc((size_t)N_NODES * 1568 * 2);   // X1_16 / X2_16 (K-padded)

    // ---- conversions / padding ----
    k_cvt_pad_cols<<<blk((long)N_NODES * 80), 256, 0, stream>>>(x, X0h, N_NODES, 78, 80);
    k_cvt_pad_rows<<<blk(80L * 780), 256, 0, stream>>>(W1, W1h, 78, 780, 80);
    k_cvt_pad_rows<<<blk(800L * 1560), 256, 0, stream>>>(W2, W2h, 780, 1560, 800);
    k_cvt_pad_rows<<<blk(1568L * 3120), 256, 0, stream>>>(W3, W3h, 1560, 3120, 1568);
    k_cvt_pad_rows<<<blk(3136L * 1024), 256, 0, stream>>>(fcg1w, g1wh, 3120, 1024, 3136);
    k_cvt_f16<<<blk(1024L * 128), 256, 0, stream>>>(fcg2w, g2wh, 1024L * 128);
    k_cvt_f16<<<blk(4224L * 1024), 256, 0, stream>>>(fc1xtw, xt1wh, 4224L * 1024);
    k_cvt_f16<<<blk(1024L * 128), 256, 0, stream>>>(fc2xtw, xt2wh, 1024L * 128);
    k_cvt_f16<<<blk(256L * 1024), 256, 0, stream>>>(fc1w, f1wh, 256L * 1024);
    k_cvt_f16<<<blk(1024L * 256), 256, 0, stream>>>(fc2w, f2wh, 1024L * 256);
    k_conv_w_reorder<<<blk(32L * 1000 * 8), 256, 0, stream>>>(cw1, cw1h, 32, 1000, 8);
    k_conv_w_reorder<<<blk(64L * 32 * 8), 256, 0, stream>>>(cw2, cw2h, 64, 32, 8);
    k_conv_w_reorder<<<blk(96L * 64 * 8), 256, 0, stream>>>(cw3, cw3h, 96, 64, 8);
    k_conv_w_reorder<<<blk(128L * 96 * 8), 256, 0, stream>>>(cw4, cw4h, 128, 96, 8);

    // ---- GAT attention pipeline ----
    auto attention = [&](const float* H, float* OUT, const float* asw, const float* adw,
                         int heads, int F) {
        long NH  = (long)N_NODES * heads;
        long EEH = (long)(N_EDGES + N_NODES) * heads;
        k_att_scores<<<blk(NH), 256, 0, stream>>>(H, asw, adw, aS, aD, heads, F);
        k_init_maxkey<<<blk(NH), 256, 0, stream>>>(mk, (int)NH);
        hipMemsetAsync(denom, 0, NH * 4, stream);
        k_edge_max<<<blk(EEH), 256, 0, stream>>>(srcIdx, dstIdx, aS, aD, mk, heads);
        k_edge_exp<<<blk(EEH), 256, 0, stream>>>(srcIdx, dstIdx, aS, aD, mk, ee, denom, heads);
        k_edge_norm<<<blk(EEH), 256, 0, stream>>>(ee, denom, dstIdx, heads);
        hipMemsetAsync(OUT, 0, (size_t)N_NODES * heads * F * 4, stream);
        long tot = (long)(N_EDGES + N_NODES) * heads * F;
        k_edge_aggr<<<blk(tot), 256, 0, stream>>>(ee, H, OUT, srcIdx, dstIdx, heads, F);
    };

    // ---- GAT layer 1: heads=10, F=78 (K padded 78->80) ----
    float* H1   = (float*)regA;
    float* OUT1 = (float*)regB;
    half_t* X1h = (half_t*)regC;
    gemm(stream, X0h, W1h, nullptr, H1, nullptr, N_NODES, 780, 80, 80, 0, 780, 780, 0, 0, 1);
    attention(H1, OUT1, as1, ad1, 10, 78);
    k_bias_act16<<<blk((long)N_NODES * 800), 256, 0, stream>>>(OUT1, b1, X1h, N_NODES, 780, 800, 2);

    // ---- GAT layer 2: heads=2, F=780 (K padded 780->800) ----
    float* H2   = (float*)regA;
    float* OUT2 = (float*)regB;
    half_t* X2h = (half_t*)regC;
    gemm(stream, X1h, W2h, nullptr, H2, nullptr, N_NODES, 1560, 800, 800, 0, 1560, 1560, 0, 0, 1);
    attention(H2, OUT2, as2, ad2, 2, 780);
    k_bias_act16<<<blk((long)N_NODES * 1568), 256, 0, stream>>>(OUT2, b2, X2h, N_NODES, 1560, 1568, 1);

    // ---- GAT layer 3: heads=1, F=3120 (K padded 1560->1568) ----
    float* H3   = (float*)regA;
    float* OUT3 = (float*)regB;
    gemm(stream, X2h, W3h, nullptr, H3, nullptr, N_NODES, 3120, 1568, 1568, 0, 3120, 3120, 0, 0, 1);
    attention(H3, OUT3, as3, ad3, 1, 3120);

    // ---- graph pool + graph-branch FCs (K padded 3120->3136) ----
    k_pool_graph<<<blk((long)N_GRAPHS * 3136), 256, 0, stream>>>(OUT3, b3, gmax16, 3120, 3136);
    gemm(stream, gmax16, g1wh, fcg1b, nullptr, g1o, N_GRAPHS, 1024, 3136, 3136, 0, 1024, 1024, 0, 1, 1);
    gemm(stream, g1o, g2wh, fcg2b, nullptr, xc, N_GRAPHS, 128, 1024, 1024, 0, 128, 256, 0, 0, 1);

    // ---- protein CNN branch (channel-last; conv1d == overlapped-row GEMM) ----
    half_t* e16 = (half_t*)regA;                              // [256, 128, 1000]
    half_t* c1  = (half_t*)regB;                              // [256, 121, 32]
    half_t* c2  = c1 + (size_t)N_GRAPHS * 121 * 32;           // [256, 114, 64]
    half_t* c3  = c2 + (size_t)N_GRAPHS * 114 * 64;           // [256, 107, 96]
    half_t* c4  = c3 + (size_t)N_GRAPHS * 107 * 96;           // [256, 100, 128]
    k_embed<<<blk((long)N_GRAPHS * 128000), 256, 0, stream>>>(tgt, emb, e16);
    gemm(stream, e16, cw1h, cb1, nullptr, c1, 121, 32, 8000, 1000, 128000, 32, 32, 121L * 32, 1, N_GRAPHS);
    gemm(stream, c1, cw2h, cb2, nullptr, c2, 114, 64, 256, 32, 121L * 32, 64, 64, 114L * 64, 1, N_GRAPHS);
    gemm(stream, c2, cw3h, cb3, nullptr, c3, 107, 96, 512, 64, 114L * 64, 96, 96, 107L * 96, 1, N_GRAPHS);
    gemm(stream, c3, cw4h, cb4, nullptr, c4, 100, 128, 768, 96, 107L * 96, 128, 128, 100L * 128, 1, N_GRAPHS);
    k_mpool_flatten<<<blk((long)N_GRAPHS * 128 * 33), 256, 0, stream>>>(c4, xt16);
    gemm(stream, xt16, xt1wh, fc1xtb, nullptr, xto1, N_GRAPHS, 1024, 4224, 4224, 0, 1024, 1024, 0, 1, 1);
    gemm(stream, xto1, xt2wh, fc2xtb, nullptr, xc + 128, N_GRAPHS, 128, 1024, 1024, 0, 128, 256, 0, 0, 1);

    // ---- fusion head ----
    gemm(stream, xc, f1wh, fc1b, nullptr, a1, N_GRAPHS, 1024, 256, 256, 0, 1024, 1024, 0, 1, 1);
    gemm(stream, a1, f2wh, fc2b, nullptr, a2, N_GRAPHS, 256, 1024, 1024, 0, 256, 256, 0, 1, 1);
    k_final<<<1, 256, 0, stream>>>(a2, outw, outb, out);
}
